// QLSTM_65481071397252
// MI455X (gfx1250) — matrix operand order repaired and checked by reading, not compiled, not run on hardware
//
#include <hip/hip_runtime.h>
#include <hip/hip_bf16.h>

// ---------------- problem constants ----------------
#define T_STEPS 512
#define B_SZ    64
#define DIN     256
#define DH      512
#define DCAT    768              // DIN + DH
#define NWG     32               // persistent workgroups
#define HPW     16               // hidden units per WG
#define COLS    64               // 4 gates * HPW weight columns per WG
#define NTHREADS 256             // 8 wave32s

// ---------------- LDS layout (dynamic) ----------------
#define OFF_WT    0                         // [COLS][DCAT] bf16 (col-major weights)
#define OFF_COMB  98304                     // [B_SZ][DCAT] bf16 ([x_t | h])
#define OFF_Z     196608                    // [B_SZ][COLS] f32
#define OFF_CST   212992                    // [B_SZ][HPW]  f32 cell state
#define OFF_BIAS  217088                    // [COLS] f32
#define SMEM_BYTES 217344

typedef __attribute__((ext_vector_type(16))) __bf16 v16bf;
typedef __attribute__((ext_vector_type(8)))  __bf16 bf16x8;
typedef __attribute__((ext_vector_type(4)))  __bf16 bf16x4;
typedef __attribute__((ext_vector_type(8)))  float  v8f;
typedef __attribute__((ext_vector_type(4)))  int    v4i;

// Address-space-qualified v4i for the async-to-LDS builtin's pointer params.
typedef __attribute__((address_space(1))) v4i gv4i;   // global (device)
typedef __attribute__((address_space(3))) v4i lv4i;   // LDS (shared)

union V16U { v16bf v; bf16x8 h[2]; };

// CDNA5 async global->LDS copy path (ASYNCcnt-tracked DMA, bypasses VGPRs).
#if defined(__AMDGCN__) && __has_builtin(__builtin_amdgcn_global_load_async_to_lds_b128) && \
    __has_builtin(__builtin_amdgcn_s_wait_asynccnt)
#define HAVE_ASYNC_LDS 1
#else
#define HAVE_ASYNC_LDS 0
#endif

__device__ __forceinline__ float sigm(float x) {
    return 1.0f / (1.0f + __expf(-x));
}

// Grid-wide arrive+spin barrier (monotonic counter; init kernel zeroes it per call).
__device__ __forceinline__ void grid_sync(unsigned* ctr, unsigned target) {
    __syncthreads();
    if (threadIdx.x == 0) {
        __threadfence();
        atomicAdd(ctr, 1u);
        while (__atomic_load_n(ctr, __ATOMIC_ACQUIRE) < target) {
            __builtin_amdgcn_s_sleep(2);
        }
    }
    __syncthreads();
}

__global__ void lstm_init(__bf16* h_ws, unsigned* ctr) {
    int i = blockIdx.x * blockDim.x + threadIdx.x;
    if (i == 0) *ctr = 0u;
    if (i < B_SZ * DH) h_ws[i] = (__bf16)0.0f;
}

__global__ __launch_bounds__(NTHREADS, 1)
void lstm_persistent(const float* __restrict__ inp,
                     const float* __restrict__ Wf, const float* __restrict__ bfp,
                     const float* __restrict__ Wi, const float* __restrict__ bip,
                     const float* __restrict__ Wg, const float* __restrict__ bgp,
                     const float* __restrict__ Wo, const float* __restrict__ bop,
                     float* __restrict__ out,
                     __bf16* __restrict__ h_ws,
                     unsigned* __restrict__ ctr)
{
    extern __shared__ char smem[];
    __bf16* wt    = (__bf16*)(smem + OFF_WT);
    __bf16* comb  = (__bf16*)(smem + OFF_COMB);
    float*  zbuf  = (float*)(smem + OFF_Z);
    float*  cst   = (float*)(smem + OFF_CST);
    float*  bgate = (float*)(smem + OFF_BIAS);

    const int tid = threadIdx.x;
    const int wg  = blockIdx.x;
    const int j0  = wg * HPW;                  // first hidden unit owned by this WG

    // ---- one-time: stage this WG's weight slice into LDS, transposed, f32->bf16.
    // wt[c][k] = W_gate(c>>4)[k][j0 + (c&15)],  c = gate*16 + u
    for (int i = tid; i < COLS * DCAT; i += NTHREADS) {
        int c = i / DCAT;
        int k = i - c * DCAT;
        const float* src = (c < 16) ? Wf : (c < 32) ? Wi : (c < 48) ? Wg : Wo;
        wt[c * DCAT + k] = (__bf16)src[k * DH + j0 + (c & 15)];
    }
    if (tid < COLS) {
        const float* bs = (tid < 16) ? bfp : (tid < 32) ? bip : (tid < 48) ? bgp : bop;
        bgate[tid] = bs[j0 + (tid & 15)];
    }
    for (int i = tid; i < B_SZ * HPW; i += NTHREADS) cst[i] = 0.0f;
    __syncthreads();

    // ---- per-wave WMMA tile assignment: 16 tiles (4 M x 4 N), 2 per wave, same M.
    const int wave = tid >> 5;
    const int lane = tid & 31;
    const int hi   = lane >> 4;                // lane half (ISA fragment layouts)
    const int l15  = lane & 15;
    const int mt   = wave >> 1;                // M tile (batch rows mt*16..+15)
    const int n0   = (wave & 1) * 2;           // N tiles n0, n0+1 (16 cols each)
    const int n1   = n0 + 1;

    float* hx_out = out + (size_t)T_STEPS * B_SZ * DH;
    float* cx_out = hx_out + (size_t)B_SZ * DH;

    // Fragment base pointers (fixed per wave; 16B-aligned b128 LDS loads).
    const __bf16* arow = comb + (mt * 16 + l15) * DCAT + 16 * hi;  // A: row = batch
    const __bf16* b0p  = wt   + (n0 * 16 + l15) * DCAT + 16 * hi;  // B: col of W
    const __bf16* b1p  = wt   + (n1 * 16 + l15) * DCAT + 16 * hi;

    for (int t = 0; t < T_STEPS; ++t) {
        // ---- stage h_{t-1} (bf16 broadcast buffer) into comb[:, 256:768].
        // Fire the DMA first so it overlaps the x_t conversion below.
        const uint4* hsrc = (const uint4*)h_ws;    // 8 bf16 per uint4
#if HAVE_ASYNC_LDS
        for (int i = tid; i < (B_SZ * DH) / 8; i += NTHREADS) {
            int b  = i >> 6;                       // 64 chunks per batch row
            int k8 = i & 63;
            __builtin_amdgcn_global_load_async_to_lds_b128(
                (gv4i*)(hsrc + i),
                (lv4i*)(comb + b * DCAT + DIN + (k8 << 3)),
                0, 0);
        }
#endif
        // ---- stage x_t (f32 -> bf16) into comb[:, 0:256] on the VALU meanwhile
        const float4* xin = (const float4*)(inp + (size_t)t * B_SZ * DIN);
        if (t + 1 < T_STEPS)
            __builtin_prefetch(inp + (size_t)(t + 1) * B_SZ * DIN, 0, 1);
        for (int i = tid; i < (B_SZ * DIN) / 4; i += NTHREADS) {
            float4 v = xin[i];
            int b = (i << 2) >> 8;
            int k = (i << 2) & 255;
            bf16x4 p = { (__bf16)v.x, (__bf16)v.y, (__bf16)v.z, (__bf16)v.w };
            *(bf16x4*)(comb + b * DCAT + k) = p;
        }
#if HAVE_ASYNC_LDS
        __builtin_amdgcn_s_wait_asynccnt(0);       // h tile landed in LDS
#else
        for (int i = tid; i < (B_SZ * DH) / 8; i += NTHREADS) {
            int b  = i >> 6;
            int k8 = i & 63;
            *(uint4*)(comb + b * DCAT + DIN + (k8 << 3)) = hsrc[i];
        }
#endif
        // barrier 1: all WGs finished READING h_ws before anyone overwrites it
        grid_sync(ctr, (unsigned)NWG * (2u * (unsigned)t + 1u));

        // ---- GEMM: z[64x64] = comb[64x768] x wt^T, bf16 WMMA, f32 accum
        v8f acc0 = {};
        v8f acc1 = {};
        for (int k0 = 0; k0 < DCAT; k0 += 32) {
            V16U a, b0, b1;
            // A 16x32 frag: lane half hi -> K 16*hi .. 16*hi+15
            a.h[0] = *(const bf16x8*)(arow + k0);
            a.h[1] = *(const bf16x8*)(arow + k0 + 8);
            // B 32x16 frag: lane = N column, elems = K 16*hi .. 16*hi+15
            b0.h[0] = *(const bf16x8*)(b0p + k0);
            b0.h[1] = *(const bf16x8*)(b0p + k0 + 8);
            b1.h[0] = *(const bf16x8*)(b1p + k0);
            b1.h[1] = *(const bf16x8*)(b1p + k0 + 8);
            acc0 = __builtin_amdgcn_wmma_f32_16x16x32_bf16(
                       false, a.v, false, b0.v, (short)0, acc0, false, false);
            acc1 = __builtin_amdgcn_wmma_f32_16x16x32_bf16(
                       false, a.v, false, b1.v, (short)0, acc1, false, false);
        }
        // ---- spill z to LDS (C/D layout: M = r + 8*hi, N = l15)
        {
            int zr = mt * 16 + 8 * hi;
            #pragma unroll
            for (int r = 0; r < 8; ++r) {
                zbuf[(zr + r) * COLS + n0 * 16 + l15] = acc0[r];
                zbuf[(zr + r) * COLS + n1 * 16 + l15] = acc1[r];
            }
        }
        __syncthreads();

        // ---- gates + state update for our 16 hidden units x 64 batch
        for (int i = tid; i < B_SZ * HPW; i += NTHREADS) {
            int b = i >> 4;
            int u = i & 15;
            float zf = zbuf[b * COLS + u]      + bgate[u];
            float zi = zbuf[b * COLS + 16 + u] + bgate[16 + u];
            float zg = zbuf[b * COLS + 32 + u] + bgate[32 + u];
            float zo = zbuf[b * COLS + 48 + u] + bgate[48 + u];
            float fg = sigm(zf);
            float ig = sigm(zi);
            float gg = tanhf(zg);
            float og = sigm(zo);
            float cn = fg * cst[i] + ig * gg;
            float hn = og * tanhf(cn);
            cst[i] = cn;
            out[(size_t)t * B_SZ * DH + b * DH + j0 + u] = hn;
            h_ws[b * DH + j0 + u] = (__bf16)hn;
            if (t == T_STEPS - 1) {
                hx_out[b * DH + j0 + u] = hn;
                cx_out[b * DH + j0 + u] = cn;
            }
        }
        // barrier 2: all h writes visible before next step's reads
        grid_sync(ctr, (unsigned)NWG * (2u * (unsigned)t + 2u));
    }
}

extern "C" void kernel_launch(void* const* d_in, const int* in_sizes, int n_in,
                              void* d_out, int out_size, void* d_ws, size_t ws_size,
                              hipStream_t stream) {
    (void)in_sizes; (void)n_in; (void)out_size; (void)ws_size;
    const float* inp = (const float*)d_in[0];
    const float* Wf  = (const float*)d_in[1];
    const float* bfp = (const float*)d_in[2];
    const float* Wi  = (const float*)d_in[3];
    const float* bip = (const float*)d_in[4];
    const float* Wg  = (const float*)d_in[5];
    const float* bgp = (const float*)d_in[6];
    const float* Wo  = (const float*)d_in[7];
    const float* bop = (const float*)d_in[8];
    float* out = (float*)d_out;

    unsigned* ctr  = (unsigned*)d_ws;                     // barrier counter
    __bf16*   h_ws = (__bf16*)((char*)d_ws + 256);        // 64x512 bf16 broadcast

    lstm_init<<<(B_SZ * DH + 255) / 256, 256, 0, stream>>>(h_ws, ctr);
    lstm_persistent<<<NWG, NTHREADS, SMEM_BYTES, stream>>>(
        inp, Wf, bfp, Wi, bip, Wg, bgp, Wo, bop, out, h_ws, ctr);
}
